// TwoLayerGraphSAGE_13005160972292
// MI455X (gfx1250) — matrix operand order repaired and checked by reading, not compiled, not run on hardware
//
#include <hip/hip_runtime.h>
#include <hip/hip_bf16.h>

// ---- types ----
typedef __attribute__((ext_vector_type(8)))  float  v8f;
typedef __attribute__((ext_vector_type(16))) __bf16 v16bf;
typedef __attribute__((ext_vector_type(8)))  __bf16 v8bf;
typedef __attribute__((ext_vector_type(4)))  __bf16 v4bf;

#define IN_DIM  128
#define HID_DIM 128
#define OUT_DIM 64
#define P_DROP  0.2f
#define KEEP_SCALE 1.25f   // 1/(1-0.2)
#define LDS_STRIDE 136     // 128 + 8 bf16 pad to avoid LDS bank conflicts

// =============== helpers ===============

__device__ __forceinline__ float elu_f(float v) {
    return v > 0.0f ? v : expm1f(v);
}

// Load one A-fragment (16x32 bf16, wave32 layout) for a given lane from an
// LDS-staged row-major [16][LDS_STRIDE] bf16 tile.
//   lanes 0-15:  M = lane,    slot s -> K = kc*32 + s
//   lanes 16-31: M = lane-16, slot s -> K = kc*32 + 16 + s
__device__ __forceinline__ v16bf load_afrag(const __bf16* tile, int lane, int kc) {
    int row  = lane & 15;
    int k0   = kc * 32 + ((lane >= 16) ? 16 : 0);
    union U { v16bf v; struct P { v8bf lo; v8bf hi; } p; } u;
    u.p.lo = *(const v8bf*)(tile + row * LDS_STRIDE + k0);
    u.p.hi = *(const v8bf*)(tile + row * LDS_STRIDE + k0 + 8);
    return u.v;
}

__device__ __forceinline__ v8f wmma_bf16(v16bf a, v16bf b, v8f c) {
    return __builtin_amdgcn_wmma_f32_16x16x32_bf16(
        /*neg_a=*/false, a, /*neg_b=*/false, b,
        /*c_mod=*/(short)0, c, /*reuse_a=*/false, /*reuse_b=*/false);
}

// =============== kernels ===============

// Zero a float region (grid-stride).
__global__ void k_zero_f32(float* __restrict__ p, long long n) {
    long long i = (long long)blockIdx.x * blockDim.x + threadIdx.x;
    long long stride = (long long)gridDim.x * blockDim.x;
    for (; i < n; i += stride) p[i] = 0.0f;
}

// Degree count: one thread per edge.
__global__ void k_count(const int* __restrict__ dst, float* __restrict__ cnt, int E) {
    int e = blockIdx.x * blockDim.x + threadIdx.x;
    if (e < E) atomicAdd(&cnt[dst[e]], 1.0f);
}

// Scatter-add f32 features: one wave per edge, 4 floats per lane (128 = 32*4).
__global__ void k_scatter_f32(const float* __restrict__ x,
                              const int* __restrict__ src,
                              const int* __restrict__ dst,
                              float* __restrict__ agg, int E) {
    int gid  = blockIdx.x * blockDim.x + threadIdx.x;
    int e    = gid >> 5;
    int lane = gid & 31;
    if (e >= E) return;
    int s = src[e], d = dst[e];
    const float4 v = *(const float4*)(x + (size_t)s * 128 + lane * 4);
    float* a = agg + (size_t)d * 128 + lane * 4;
    atomicAdd(a + 0, v.x);
    atomicAdd(a + 1, v.y);
    atomicAdd(a + 2, v.z);
    atomicAdd(a + 3, v.w);
}

// Scatter-add bf16 features (layer-2 hidden state): one wave per edge.
__global__ void k_scatter_bf16(const __bf16* __restrict__ h,
                               const int* __restrict__ src,
                               const int* __restrict__ dst,
                               float* __restrict__ agg, int E) {
    int gid  = blockIdx.x * blockDim.x + threadIdx.x;
    int e    = gid >> 5;
    int lane = gid & 31;
    if (e >= E) return;
    int s = src[e], d = dst[e];
    const v4bf v = *(const v4bf*)(h + (size_t)s * 128 + lane * 4);
    float* a = agg + (size_t)d * 128 + lane * 4;
    atomicAdd(a + 0, (float)v[0]);
    atomicAdd(a + 1, (float)v[1]);
    atomicAdd(a + 2, (float)v[2]);
    atomicAdd(a + 3, (float)v[3]);
}

// Pre-swizzle a row-major f32 weight [128 x ncols] into WMMA B-fragment order:
// frag index = (nt*4 + kc)*32 + lane ; 16 contiguous bf16 per lane.
//   N = nt*16 + lane%16 ;  K(slot s) = kc*32 + (lane>=16 ? 16 : 0) + s
__global__ void k_prep_w(const float* __restrict__ W, __bf16* __restrict__ out, int ncols) {
    int idx  = blockIdx.x * blockDim.x + threadIdx.x;   // one thread per (nt,kc,lane)
    int ntk  = idx >> 5;
    int lane = idx & 31;
    int kc   = ntk & 3;
    int nt   = ntk >> 2;
    if (nt * 16 >= ncols) return;
    int n     = nt * 16 + (lane & 15);
    int kbase = kc * 32 + ((lane >= 16) ? 16 : 0);
    __bf16* o = out + (size_t)idx * 16;
#pragma unroll
    for (int s = 0; s < 16; ++s)
        o[s] = (__bf16)W[(size_t)(kbase + s) * ncols + n];
}

// Layer-1 node update: h = dropout(elu(mean_agg @ W1l + x @ W1r + b1)) -> bf16
// Block = 128 threads (4 waves), one 16-node tile per block.
__global__ void k_node1(const float* __restrict__ x,
                        const float* __restrict__ agg,
                        const float* __restrict__ cnt,
                        const __bf16* __restrict__ wl,   // prepped frags, 8 N-tiles
                        const __bf16* __restrict__ wr,
                        const float* __restrict__ b1,
                        const float* __restrict__ drop_u,
                        __bf16* __restrict__ h, int N) {
    __shared__ alignas(16) __bf16 As[16 * LDS_STRIDE];  // mean-aggregated rows
    __shared__ alignas(16) __bf16 Xs[16 * LDS_STRIDE];  // root rows

    const int row0 = blockIdx.x * 16;
    const int t    = threadIdx.x;
    // stage: thread t -> row t/8, cols (t%8)*16 .. +15
    {
        int r    = t >> 3;
        int c0   = (t & 7) * 16;
        int node = row0 + r;
        if (node < N) {
            float inv = 1.0f / fmaxf(cnt[node], 1.0f);
            const float* ar = agg + (size_t)node * 128 + c0;
            const float* xr = x   + (size_t)node * 128 + c0;
#pragma unroll
            for (int i = 0; i < 16; ++i) {
                As[r * LDS_STRIDE + c0 + i] = (__bf16)(ar[i] * inv);
                Xs[r * LDS_STRIDE + c0 + i] = (__bf16)(xr[i]);
            }
        } else {
#pragma unroll
            for (int i = 0; i < 16; ++i) {
                As[r * LDS_STRIDE + c0 + i] = (__bf16)0.0f;
                Xs[r * LDS_STRIDE + c0 + i] = (__bf16)0.0f;
            }
        }
    }
    __syncthreads();

    const int lane = t & 31;
    const int wave = t >> 5;

#pragma unroll
    for (int rep = 0; rep < 2; ++rep) {
        const int nt = wave + rep * 4;          // 8 N-tiles of HID=128
        v8f acc = {};
#pragma unroll
        for (int kc = 0; kc < 4; ++kc) {        // K = 128 = 4 * 32
            v16bf aA = load_afrag(As, lane, kc);
            v16bf aX = load_afrag(Xs, lane, kc);
            v16bf bL = ((const v16bf*)wl)[(nt * 4 + kc) * 32 + lane];
            v16bf bR = ((const v16bf*)wr)[(nt * 4 + kc) * 32 + lane];
            acc = wmma_bf16(aA, bL, acc);
            acc = wmma_bf16(aX, bR, acc);
        }
        // epilogue: C/D layout -> reg r: row = r + (lane<16 ? 0 : 8), col = nt*16 + lane%16
        const int col   = nt * 16 + (lane & 15);
        const int rbase = (lane < 16) ? 0 : 8;
        const float bias = b1[col];
#pragma unroll
        for (int rr = 0; rr < 8; ++rr) {
            int node = row0 + rbase + rr;
            if (node < N) {
                float v = elu_f(acc[rr] + bias);
                float keep = (drop_u[(size_t)node * 128 + col] >= P_DROP) ? KEEP_SCALE : 0.0f;
                h[(size_t)node * 128 + col] = (__bf16)(v * keep);
            }
        }
    }
}

// Layer-2 node update: out = elu(mean_agg2 @ W2l + h @ W2r + b2) -> f32 [N,64]
__global__ void k_node2(const __bf16* __restrict__ h,
                        const float* __restrict__ agg,
                        const float* __restrict__ cnt,
                        const __bf16* __restrict__ wl,   // prepped frags, 4 N-tiles
                        const __bf16* __restrict__ wr,
                        const float* __restrict__ b2,
                        float* __restrict__ out, int N) {
    __shared__ alignas(16) __bf16 As[16 * LDS_STRIDE];
    __shared__ alignas(16) __bf16 Hs[16 * LDS_STRIDE];

    const int row0 = blockIdx.x * 16;
    const int t    = threadIdx.x;
    {
        int r    = t >> 3;
        int c0   = (t & 7) * 16;
        int node = row0 + r;
        if (node < N) {
            float inv = 1.0f / fmaxf(cnt[node], 1.0f);
            const float*  ar = agg + (size_t)node * 128 + c0;
            const __bf16* hr = h   + (size_t)node * 128 + c0;
#pragma unroll
            for (int i = 0; i < 16; ++i) {
                As[r * LDS_STRIDE + c0 + i] = (__bf16)(ar[i] * inv);
                Hs[r * LDS_STRIDE + c0 + i] = hr[i];
            }
        } else {
#pragma unroll
            for (int i = 0; i < 16; ++i) {
                As[r * LDS_STRIDE + c0 + i] = (__bf16)0.0f;
                Hs[r * LDS_STRIDE + c0 + i] = (__bf16)0.0f;
            }
        }
    }
    __syncthreads();

    const int lane = t & 31;
    const int nt   = t >> 5;                    // 4 N-tiles of OUT=64
    v8f acc = {};
#pragma unroll
    for (int kc = 0; kc < 4; ++kc) {
        v16bf aA = load_afrag(As, lane, kc);
        v16bf aH = load_afrag(Hs, lane, kc);
        v16bf bL = ((const v16bf*)wl)[(nt * 4 + kc) * 32 + lane];
        v16bf bR = ((const v16bf*)wr)[(nt * 4 + kc) * 32 + lane];
        acc = wmma_bf16(aA, bL, acc);
        acc = wmma_bf16(aH, bR, acc);
    }
    const int col   = nt * 16 + (lane & 15);
    const int rbase = (lane < 16) ? 0 : 8;
    const float bias = b2[col];
#pragma unroll
    for (int rr = 0; rr < 8; ++rr) {
        int node = row0 + rbase + rr;
        if (node < N) {
            out[(size_t)node * OUT_DIM + col] = elu_f(acc[rr] + bias);
        }
    }
}

// =============== launcher ===============

extern "C" void kernel_launch(void* const* d_in, const int* in_sizes, int n_in,
                              void* d_out, int out_size, void* d_ws, size_t ws_size,
                              hipStream_t stream) {
    const float* x      = (const float*)d_in[0];
    const int*   eidx   = (const int*)  d_in[1];
    const float* drop_u = (const float*)d_in[2];
    const float* W1l    = (const float*)d_in[3];
    const float* W1r    = (const float*)d_in[4];
    const float* b1     = (const float*)d_in[5];
    const float* W2l    = (const float*)d_in[6];
    const float* W2r    = (const float*)d_in[7];
    const float* b2     = (const float*)d_in[8];
    float* out = (float*)d_out;

    const int N = in_sizes[0] / IN_DIM;     // 100000
    const int E = in_sizes[1] / 2;          // 1600000
    const int* src = eidx;
    const int* dst = eidx + E;

    // workspace layout (bytes)
    char* ws = (char*)d_ws;
    const size_t aggBytes = (size_t)N * 128 * sizeof(float);       // 51.2 MB
    float*  agg  = (float*)(ws);                                   // reused for both layers
    float*  cnt  = (float*)(ws + aggBytes);                        // N floats
    size_t  off  = aggBytes + (size_t)N * sizeof(float);
    __bf16* hbuf = (__bf16*)(ws + off);                            // N*128 bf16
    off += (size_t)N * 128 * sizeof(__bf16);
    __bf16* w1l_p = (__bf16*)(ws + off); off += (size_t)128 * 128 * sizeof(__bf16);
    __bf16* w1r_p = (__bf16*)(ws + off); off += (size_t)128 * 128 * sizeof(__bf16);
    __bf16* w2l_p = (__bf16*)(ws + off); off += (size_t)128 * 64  * sizeof(__bf16);
    __bf16* w2r_p = (__bf16*)(ws + off);

    const int tiles = (N + 15) / 16;

    // 0) pre-swizzle weights into WMMA B-fragment order
    k_prep_w<<<(1024 + 255) / 256, 256, 0, stream>>>(W1l, w1l_p, HID_DIM);
    k_prep_w<<<(1024 + 255) / 256, 256, 0, stream>>>(W1r, w1r_p, HID_DIM);
    k_prep_w<<<(512  + 255) / 256, 256, 0, stream>>>(W2l, w2l_p, OUT_DIM);
    k_prep_w<<<(512  + 255) / 256, 256, 0, stream>>>(W2r, w2r_p, OUT_DIM);

    // 1) zero agg + cnt (contiguous), count degrees, scatter x
    {
        long long nz = (long long)N * 128 + N;
        k_zero_f32<<<2048, 256, 0, stream>>>(agg, nz);
    }
    k_count<<<(E + 255) / 256, 256, 0, stream>>>(dst, cnt, E);
    {
        long long threads = (long long)E * 32;
        k_scatter_f32<<<(unsigned)((threads + 255) / 256), 256, 0, stream>>>(x, src, dst, agg, E);
    }

    // 2) layer-1 node update (WMMA) -> h (bf16)
    k_node1<<<tiles, 128, 0, stream>>>(x, agg, cnt, w1l_p, w1r_p, b1, drop_u, hbuf, N);

    // 3) re-zero agg, scatter h, layer-2 node update -> out (f32)
    k_zero_f32<<<2048, 256, 0, stream>>>(agg, (long long)N * 128);
    {
        long long threads = (long long)E * 32;
        k_scatter_bf16<<<(unsigned)((threads + 255) / 256), 256, 0, stream>>>(hbuf, src, dst, agg, E);
    }
    k_node2<<<tiles, 128, 0, stream>>>(hbuf, agg, cnt, w2l_p, w2r_p, b2, out, N);
}
